// SPINN_46961172415170
// MI455X (gfx1250) — matrix operand order repaired and checked by reading, not compiled, not run on hardware
//
#include <hip/hip_runtime.h>
#include <math.h>

// Problem constants (from reference)
#define B_    256
#define L_    128
#define D_    256
#define TS_   128
#define T_    255
#define MAXS_ 130          // L + 2 sentinel slots
#define KT_   896          // 3*D + TS  (tracker GEMM K)
#define NT_   512          // 4*TS      (tracker GEMM N)
#define KR_   640          // 2*D + TS  (reduce GEMM K)
#define NR_   1280         // 5*D       (reduce GEMM N)

typedef __bf16 bf16;
typedef __attribute__((ext_vector_type(16))) __bf16 v16bf;
typedef __attribute__((ext_vector_type(8)))  __bf16 v8bf;
typedef __attribute__((ext_vector_type(8)))  float  v8f;

__device__ __forceinline__ float sigf(float x) { return 1.0f / (1.0f + __expf(-x)); }

__device__ __forceinline__ v16bf cat16(v8bf lo, v8bf hi) {
    return __builtin_shufflevector(lo, hi,
        0, 1, 2, 3, 4, 5, 6, 7, 8, 9, 10, 11, 12, 13, 14, 15);
}

// ---------------------------------------------------------------------------
// One-time init: zero stacks + tracker state, set ptr/bptr (parity-0 copy)
// ---------------------------------------------------------------------------
__global__ void k_init(float* __restrict__ stack_c, float* __restrict__ stack_h,
                       float* __restrict__ tc, float* __restrict__ th,
                       int* __restrict__ ptrA, int* __restrict__ bptrA) {
    size_t i = (size_t)blockIdx.x * blockDim.x + threadIdx.x;
    size_t stride = (size_t)gridDim.x * blockDim.x;
    const size_t NS = (size_t)B_ * MAXS_ * D_;
    for (size_t k = i; k < NS; k += stride) { stack_c[k] = 0.f; stack_h[k] = 0.f; }
    const size_t NTR = (size_t)B_ * TS_;
    for (size_t k = i; k < NTR; k += stride) { tc[k] = 0.f; th[k] = 0.f; }
    if (i < B_) { ptrA[i] = 2; bptrA[i] = L_; }
}

// ---------------------------------------------------------------------------
// One-time weight conversion fp32 -> bf16, concatenated row blocks
// Wt rows: [W_buf(256); W_s1(256); W_s2(256); W_lat(128)]  -> [896 x 512]
// ---------------------------------------------------------------------------
__global__ void k_conv_wt(const float* __restrict__ Wbuf, const float* __restrict__ Ws1,
                          const float* __restrict__ Ws2, const float* __restrict__ Wlat,
                          bf16* __restrict__ Wt) {
    int i = blockIdx.x * blockDim.x + threadIdx.x;
    if (i >= KT_ * NT_) return;
    int r = i / NT_, n = i % NT_;
    float v;
    if (r < 256)      v = Wbuf[r * NT_ + n];
    else if (r < 512) v = Ws1[(r - 256) * NT_ + n];
    else if (r < 768) v = Ws2[(r - 512) * NT_ + n];
    else              v = Wlat[(r - 768) * NT_ + n];
    Wt[i] = (bf16)v;
}

// Wr rows: [W_left(256); W_right(256); W_track(128)] -> [640 x 1280]
__global__ void k_conv_wr(const float* __restrict__ Wleft, const float* __restrict__ Wright,
                          const float* __restrict__ Wtrack, bf16* __restrict__ Wr) {
    int i = blockIdx.x * blockDim.x + threadIdx.x;
    if (i >= KR_ * NR_) return;
    int r = i / NR_, n = i % NR_;
    float v;
    if (r < 256)      v = Wleft[r * NR_ + n];
    else if (r < 512) v = Wright[(r - 256) * NR_ + n];
    else              v = Wtrack[(r - 512) * NR_ + n];
    Wr[i] = (bf16)v;
}

// ---------------------------------------------------------------------------
// Per-step: gather tracker A-panel [B x 896] bf16
// layout: [top_buf_h(256) | s1_h(256) | s2_h(256) | th(128)]
// ---------------------------------------------------------------------------
__global__ void k_gather_t(const float* __restrict__ buf_h, const float* __restrict__ stack_h,
                           const float* __restrict__ th,
                           const int* __restrict__ ptrI, const int* __restrict__ bptrI,
                           bf16* __restrict__ At) {
    int i = blockIdx.x * blockDim.x + threadIdx.x;
    if (i >= B_ * KT_) return;
    int b = i / KT_, k = i % KT_;
    float v;
    if (k < D_) {
        int bp = bptrI[b];
        v = (bp == 0) ? 0.f : buf_h[((size_t)b * L_ + (bp - 1)) * D_ + k];
    } else if (k < 2 * D_) {
        int p = ptrI[b];
        v = stack_h[((size_t)b * MAXS_ + (p - 1)) * D_ + (k - D_)];
    } else if (k < 3 * D_) {
        int p = ptrI[b];
        v = stack_h[((size_t)b * MAXS_ + (p - 2)) * D_ + (k - 2 * D_)];
    } else {
        v = th[b * TS_ + (k - 3 * D_)];
    }
    At[(size_t)b * KT_ + k] = (bf16)v;
}

// ---------------------------------------------------------------------------
// bf16 WMMA GEMM: C[MxN] = A[MxK] * B[KxN], f32 accumulate.
// Workgroup = 128 threads (4 waves) -> 32(M) x 64(N) macro-tile.
// Wave (wm,wn) computes a 16x32 strip = 2 WMMA tiles sharing one A fragment.
// B tile [32K x 64N] is double-buffered in LDS.  Steady-state body is
// branch-free: A loads -> LDS B loads -> prefetch next B tile (globals issued
// AFTER the consumed loads so WMMA can proceed at loadcnt<=2 with the
// prefetch in flight) -> 2 WMMAs -> publish tile -> ONE barrier.  Last K-step
// is peeled.  Fragment layouts per CDNA5 ISA 7.12.2.  All dims divide
// exactly -> no divergent guards, EXEC stays all-ones for WMMA.
// ---------------------------------------------------------------------------
__global__ __launch_bounds__(128) void k_gemm_bf16(const bf16* __restrict__ A,
                                                   const bf16* __restrict__ Bm,
                                                   float* __restrict__ C,
                                                   int K, int N) {
    __shared__ bf16 Bs[2][32 * 64];              // 2 x 4 KB staged B tiles
    const int lane  = threadIdx.x & 31;
    const int wave  = threadIdx.x >> 5;
    const int wgN   = N >> 6;                    // workgroup columns
    const int m0    = (blockIdx.x / wgN) * 32 + (wave >> 1) * 16;
    const int nBase = (blockIdx.x % wgN) * 64;
    const int nOff  = (wave & 1) * 32;           // this wave's 32-col strip in LDS
    const int n0    = nBase + nOff;

    // A fragment addressing: lanes 0-15 -> row m0+lane,    K chunks {0..7,16..23}
    //                        lanes 16-31 -> row m0+lane-16, K chunks {8..15,24..31}
    const int  aRow = m0 + (lane & 15);
    const int  aK0  = (lane < 16) ? 0 : 8;
    const bf16* aPtr = A + (size_t)aRow * K + aK0;

    // B staging: 32 rows x 64 cols = 256 chunks of 8 bf16; each thread does 2
    // (rows c0r and c0r+16, same 8-col chunk).
    const int c0r = threadIdx.x >> 3;
    const int c0c = (threadIdx.x & 7) * 8;
    const int c1r = c0r + 16;
    const bf16* gB0 = Bm + (size_t)c0r * N + nBase + c0c;
    const bf16* gB1 = Bm + (size_t)c1r * N + nBase + c0c;
    const int   ls0 = c0r * 64 + c0c;
    const int   ls1 = c1r * 64 + c0c;

    // prologue: stage tile 0 into buffer 0
    *(v8bf*)(&Bs[0][ls0]) = *(const v8bf*)(gB0);
    *(v8bf*)(&Bs[0][ls1]) = *(const v8bf*)(gB1);
    __syncthreads();

    v8f acc0 = {}, acc1 = {};
    int p = 0;
    const int steps = K >> 5;

    // steady state: branch-free pipelined body, last iteration peeled
    for (int s = 0; s < steps - 1; ++s) {
        const int k0 = s << 5;
        // consumed loads first (A globals, B from LDS) ...
        v16bf af = cat16(*(const v8bf*)(aPtr + k0), *(const v8bf*)(aPtr + k0 + 16));
        const bf16* bl = &Bs[p][(lane >> 4) * 8 * 64 + nOff + (lane & 15)];
        v16bf b0, b1;
#pragma unroll
        for (int e = 0; e < 16; ++e) {
            const int kk = (e & 7) + ((e >> 3) << 4);
            b0[e] = bl[kk * 64];
            b1[e] = bl[kk * 64 + 16];
        }
        // ... then the next-tile prefetch, so it stays in flight across WMMA
        v8bf nb0 = *(const v8bf*)(gB0 + (size_t)(k0 + 32) * N);
        v8bf nb1 = *(const v8bf*)(gB1 + (size_t)(k0 + 32) * N);

        acc0 = __builtin_amdgcn_wmma_f32_16x16x32_bf16(
                   false, af, false, b0, (short)0, acc0, false, false);
        acc1 = __builtin_amdgcn_wmma_f32_16x16x32_bf16(
                   false, af, false, b1, (short)0, acc1, false, false);

        *(v8bf*)(&Bs[p ^ 1][ls0]) = nb0;         // publish next tile
        *(v8bf*)(&Bs[p ^ 1][ls1]) = nb1;
        __syncthreads();
        p ^= 1;
    }

    // peeled tail: consume last tile, no prefetch / store / barrier
    {
        const int k0 = (steps - 1) << 5;
        v16bf af = cat16(*(const v8bf*)(aPtr + k0), *(const v8bf*)(aPtr + k0 + 16));
        const bf16* bl = &Bs[p][(lane >> 4) * 8 * 64 + nOff + (lane & 15)];
        v16bf b0, b1;
#pragma unroll
        for (int e = 0; e < 16; ++e) {
            const int kk = (e & 7) + ((e >> 3) << 4);
            b0[e] = bl[kk * 64];
            b1[e] = bl[kk * 64 + 16];
        }
        acc0 = __builtin_amdgcn_wmma_f32_16x16x32_bf16(
                   false, af, false, b0, (short)0, acc0, false, false);
        acc1 = __builtin_amdgcn_wmma_f32_16x16x32_bf16(
                   false, af, false, b1, (short)0, acc1, false, false);
    }

    // C/D layout: VGPR r -> row m0 + r + 8*(lane>=16), col n0 + (lane&15)
    const int cRow = m0 + ((lane >> 4) << 3);
    float* cp = C + (size_t)cRow * N + n0 + (lane & 15);
#pragma unroll
    for (int r = 0; r < 8; ++r) cp[(size_t)r * N] = acc0[r];
#pragma unroll
    for (int r = 0; r < 8; ++r) cp[(size_t)r * N + 16] = acc1[r];
}

// ---------------------------------------------------------------------------
// Per-step: tracker LSTM update + build reduce A-panel [B x 640] bf16
// Ar layout: [s2_h(256) | s1_h(256) | th_new(128)]  (s2/s1 reused from At)
// ---------------------------------------------------------------------------
__global__ void k_track_upd(const float* __restrict__ lstm, const float* __restrict__ b_lat,
                            float* __restrict__ tc, float* __restrict__ th,
                            const bf16* __restrict__ At, bf16* __restrict__ Ar, int t) {
    int i = blockIdx.x * blockDim.x + threadIdx.x;    // B * 512
    if (i >= B_ * 512) return;
    int b = i >> 9, j = i & 511;
    // copy already-bf16 s2 (At[512..767]) then s1 (At[256..511]) into Ar[0..511]
    bf16 cv = (j < 256) ? At[(size_t)b * KT_ + 512 + j] : At[(size_t)b * KT_ + j];
    Ar[(size_t)b * KR_ + j] = cv;
    if (j < TS_) {
        const float* lb = lstm + (size_t)b * NT_;
        float a  = lb[j];
        float ii = lb[TS_ + j];
        float f  = lb[2 * TS_ + j];
        float o  = lb[3 * TS_ + j];
        if (t > 0) {
            a  += b_lat[j];
            ii += b_lat[TS_ + j];
            f  += b_lat[2 * TS_ + j];
            o  += b_lat[3 * TS_ + j];
        }
        float c = tanhf(a) * sigf(ii) + sigf(f) * tc[b * TS_ + j];
        float h = sigf(o) * tanhf(c);
        tc[b * TS_ + j] = c;
        th[b * TS_ + j] = h;
        Ar[(size_t)b * KR_ + 2 * D_ + j] = (bf16)h;
    }
}

// ---------------------------------------------------------------------------
// Per-step: TreeLSTM reduce + shift/reduce select + stack write + ptr update
// ptr/bptr double-buffered (in -> out) so reads never race the j==0 writer.
// ---------------------------------------------------------------------------
__global__ void k_reduce_upd(const float* __restrict__ rin, const float* __restrict__ b_left,
                             const float* __restrict__ buf_c, const float* __restrict__ buf_h,
                             float* __restrict__ stack_c, float* __restrict__ stack_h,
                             const int* __restrict__ ptrI, const int* __restrict__ bptrI,
                             int* __restrict__ ptrO, int* __restrict__ bptrO,
                             const int* __restrict__ trans, int t) {
    int i = blockIdx.x * blockDim.x + threadIdx.x;    // B * D
    if (i >= B_ * D_) return;
    int b = i >> 8, j = i & 255;
    int p  = ptrI[b];
    int bp = bptrI[b];
    const float* rb = rin + (size_t)b * NR_;
    float a2 = rb[j]           + b_left[j];
    float i2 = rb[D_ + j]      + b_left[D_ + j];
    float f1 = rb[2 * D_ + j]  + b_left[2 * D_ + j];
    float f2 = rb[3 * D_ + j]  + b_left[3 * D_ + j];
    float o2 = rb[4 * D_ + j]  + b_left[4 * D_ + j];
    float s2c = stack_c[((size_t)b * MAXS_ + (p - 2)) * D_ + j];
    float s1c = stack_c[((size_t)b * MAXS_ + (p - 1)) * D_ + j];
    float rc = tanhf(a2) * sigf(i2) + sigf(f1) * s2c + sigf(f2) * s1c;
    float rh = sigf(o2) * tanhf(rc);
    bool sh = (trans[(size_t)b * T_ + t] == 0);       // T_SHIFT == 0
    float tbc = 0.f, tbh = 0.f;
    if (bp > 0) {
        tbc = buf_c[((size_t)b * L_ + (bp - 1)) * D_ + j];
        tbh = buf_h[((size_t)b * L_ + (bp - 1)) * D_ + j];
    }
    int wp = sh ? p : (p - 2);
    stack_c[((size_t)b * MAXS_ + wp) * D_ + j] = sh ? tbc : rc;
    stack_h[((size_t)b * MAXS_ + wp) * D_ + j] = sh ? tbh : rh;
    if (j == 0) {
        ptrO[b]  = sh ? (p + 1) : (p - 1);
        bptrO[b] = sh ? (bp - 1) : bp;
    }
}

// ---------------------------------------------------------------------------
// Final: out[b] = concat(stack_h[b][ptr-1], stack_c[b][ptr-1])  -> [B x 2D]
// ---------------------------------------------------------------------------
__global__ void k_final(const float* __restrict__ stack_c, const float* __restrict__ stack_h,
                        const int* __restrict__ ptrI, float* __restrict__ out) {
    int i = blockIdx.x * blockDim.x + threadIdx.x;    // B * 2D
    if (i >= B_ * 2 * D_) return;
    int b = i >> 9, j = i & 511;
    int p = ptrI[b];
    float v = (j < D_) ? stack_h[((size_t)b * MAXS_ + (p - 1)) * D_ + j]
                       : stack_c[((size_t)b * MAXS_ + (p - 1)) * D_ + (j - D_)];
    out[i] = v;
}

// ---------------------------------------------------------------------------
// Host-side orchestration
// ---------------------------------------------------------------------------
extern "C" void kernel_launch(void* const* d_in, const int* in_sizes, int n_in,
                              void* d_out, int out_size, void* d_ws, size_t ws_size,
                              hipStream_t stream) {
    const float* buf_c   = (const float*)d_in[0];
    const float* buf_h   = (const float*)d_in[1];
    const float* W_buf   = (const float*)d_in[2];
    const float* W_s1    = (const float*)d_in[3];
    const float* W_s2    = (const float*)d_in[4];
    const float* W_lat   = (const float*)d_in[5];
    const float* b_lat   = (const float*)d_in[6];
    const float* W_left  = (const float*)d_in[7];
    const float* b_left  = (const float*)d_in[8];
    const float* W_right = (const float*)d_in[9];
    const float* W_track = (const float*)d_in[10];
    const int*   trans   = (const int*)d_in[11];
    float* out = (float*)d_out;

    // workspace layout (all offsets 1KB-aligned)
    char* ws = (char*)d_ws;
    bf16*  Wt      = (bf16*)(ws + 0);                       //  896*512*2  = 917504
    bf16*  Wr      = (bf16*)(ws + 917504);                  //  640*1280*2 = 1638400
    bf16*  At      = (bf16*)(ws + 2555904);                 //  256*896*2  = 458752
    bf16*  Ar      = (bf16*)(ws + 3014656);                 //  256*640*2  = 327680
    float* lstm    = (float*)(ws + 3342336);                //  256*512*4  = 524288
    float* rin     = (float*)(ws + 3866624);                //  256*1280*4 = 1310720
    float* tc      = (float*)(ws + 5177344);                //  256*128*4  = 131072
    float* th      = (float*)(ws + 5308416);                //  256*128*4  = 131072
    int*   ptrA    = (int*)(ws + 5439488);                  //  1024
    int*   ptrB    = (int*)(ws + 5440512);                  //  1024
    int*   bptrA   = (int*)(ws + 5441536);                  //  1024
    int*   bptrB   = (int*)(ws + 5442560);                  //  1024
    float* stack_c = (float*)(ws + 5443584);                //  256*130*256*4 = 34078720
    float* stack_h = (float*)(ws + 39522304);               //  34078720 (end 73601024)

    // init + weight conversion (state re-created every call => replay-safe)
    k_init<<<1024, 256, 0, stream>>>(stack_c, stack_h, tc, th, ptrA, bptrA);
    k_conv_wt<<<(KT_ * NT_ + 255) / 256, 256, 0, stream>>>(W_buf, W_s1, W_s2, W_lat, Wt);
    k_conv_wr<<<(KR_ * NR_ + 255) / 256, 256, 0, stream>>>(W_left, W_right, W_track, Wr);

    const int wgT = (B_ / 32) * (NT_ / 64);   // 8 * 8  = 64 workgroups
    const int wgR = (B_ / 32) * (NR_ / 64);   // 8 * 20 = 160 workgroups

    for (int t = 0; t < T_; ++t) {
        int*  pI = (t & 1) ? ptrB : ptrA;
        int*  pO = (t & 1) ? ptrA : ptrB;
        int* bpI = (t & 1) ? bptrB : bptrA;
        int* bpO = (t & 1) ? bptrA : bptrB;

        k_gather_t<<<(B_ * KT_ + 255) / 256, 256, 0, stream>>>(
            buf_h, stack_h, th, pI, bpI, At);
        k_gemm_bf16<<<wgT, 128, 0, stream>>>(At, Wt, lstm, KT_, NT_);
        k_track_upd<<<(B_ * 512 + 255) / 256, 256, 0, stream>>>(
            lstm, b_lat, tc, th, At, Ar, t);
        k_gemm_bf16<<<wgR, 128, 0, stream>>>(Ar, Wr, rin, KR_, NR_);
        k_reduce_upd<<<(B_ * D_ + 255) / 256, 256, 0, stream>>>(
            rin, b_left, buf_c, buf_h, stack_c, stack_h, pI, bpI, pO, bpO, trans, t);
    }

    // after step T-1 (t=254), final ptr lives in parity (T_ & 1) == 1 -> ptrB
    k_final<<<(B_ * 2 * D_ + 255) / 256, 256, 0, stream>>>(stack_c, stack_h, ptrB, out);
    (void)in_sizes; (void)n_in; (void)out_size; (void)ws_size;
}
